// PlugableIPAdapter_15410342658191
// MI455X (gfx1250) — compile-verified, matrix operand order checked by reading, not run on hardware
//
#include <hip/hip_runtime.h>
#include <hip/hip_bf16.h>

typedef unsigned short ushort_t;
typedef __attribute__((ext_vector_type(16))) __bf16 v16bf;
typedef __attribute__((ext_vector_type(8)))  float  v8f;

// ---------- helpers ----------
__device__ inline ushort_t f2bf(float x) {
  union { float f; unsigned u; } c; c.f = x;
  unsigned u = c.u;
  u += 0x7fffu + ((u >> 16) & 1u);   // round-to-nearest-even
  return (ushort_t)(u >> 16);
}

// Build a 16xbf16 A/B fragment from two contiguous 16-byte chunks
// (ISA 7.12.2: vgpr0-3 = K 8*hi..8*hi+7, vgpr4-7 = K 16+8*hi..16+8*hi+7).
__device__ inline v16bf frag_ld(const ushort_t* lo, const ushort_t* hi) {
  union { uint4 u[2]; v16bf v; } t;
  t.u[0] = *(const uint4*)lo;
  t.u[1] = *(const uint4*)hi;
  return t.v;
}

__device__ inline v8f wmma_bf16(v16bf a, v16bf b, v8f c) {
  return __builtin_amdgcn_wmma_f32_16x16x32_bf16(false, a, false, b, (short)0, c, false, false);
}

// Async global->LDS copy, 16 B per lane (VGLOBAL GV mode, tracked by ASYNCcnt).
__device__ inline void async_ld_b128(unsigned lds_off, const void* gaddr) {
  asm volatile("global_load_async_to_lds_b128 %0, %1, off"
               :: "v"(lds_off), "v"(gaddr) : "memory");
}
__device__ inline void wait_async0() {
  asm volatile("s_wait_asynccnt 0x0" ::: "memory");
}

// ---------- elementwise f32 -> bf16 (8 elems/thread) ----------
__global__ __launch_bounds__(256)
void cvt_f32_bf16_kernel(const float* __restrict__ in, ushort_t* __restrict__ out, int n) {
  int i = (blockIdx.x * blockDim.x + threadIdx.x) * 8;
  if (i >= n) return;
  float4 a = *(const float4*)(in + i);
  float4 b = *(const float4*)(in + i + 4);
  uint4 p;
  p.x = (unsigned)f2bf(a.x) | ((unsigned)f2bf(a.y) << 16);
  p.y = (unsigned)f2bf(a.z) | ((unsigned)f2bf(a.w) << 16);
  p.z = (unsigned)f2bf(b.x) | ((unsigned)f2bf(b.y) << 16);
  p.w = (unsigned)f2bf(b.z) | ((unsigned)f2bf(b.w) << 16);
  *(uint4*)(out + i) = p;
}

// ---------- W[K,N] f32  ->  Wt[N,K] bf16 (tiled transpose) ----------
__global__ __launch_bounds__(256)
void wt_transpose_cvt_kernel(const float* __restrict__ W, ushort_t* __restrict__ Wt,
                             int K, int N) {
  __shared__ ushort_t tile[32][33];
  int kb = blockIdx.y * 32, nb = blockIdx.x * 32;
  int tx = threadIdx.x, ty = threadIdx.y;           // (32, 8)
  #pragma unroll
  for (int i = 0; i < 4; i++) {
    int k = kb + ty + i * 8;
    tile[ty + i * 8][tx] = f2bf(W[(size_t)k * N + nb + tx]);
  }
  __syncthreads();
  #pragma unroll
  for (int i = 0; i < 4; i++) {
    int n = nb + ty + i * 8;
    Wt[(size_t)n * K + kb + tx] = tile[tx][ty + i * 8];
  }
}

// ---------- small GEMM (bounds-checked): out_bf16[M,N] = A_f32[M,K] * W_f32[K,N] ----------
// Block 128 threads (4 waves), tile 64x64, K step 32. Used for the tiny K/V projections.
__global__ __launch_bounds__(128)
void gemm_small_wmma(const float* __restrict__ Af, const float* __restrict__ W,
                     ushort_t* __restrict__ Ob, int M, int K, int N) {
  __shared__ ushort_t As[64 * 32];
  __shared__ ushort_t BsT[64 * 32];
  const int tid = threadIdx.x;
  const int wv = tid >> 5, lane = tid & 31;
  const int ln = lane & 15, hi = lane >> 4;
  const int m0 = blockIdx.y * 64, n0 = blockIdx.x * 64;

  v8f zero = {};
  v8f acc[4] = {zero, zero, zero, zero};

  for (int k0 = 0; k0 < K; k0 += 32) {
    #pragma unroll
    for (int it = 0; it < 4; it++) {
      int i = tid + it * 128;            // 0..511
      int r = i >> 3;                    // 0..63
      int k4 = (i & 7) << 2;             // 0..28
      int gr = m0 + r;
      ushort_t h0 = 0, h1 = 0, h2 = 0, h3 = 0;
      if (gr < M) {
        float4 f = *(const float4*)(Af + (size_t)gr * K + k0 + k4);
        h0 = f2bf(f.x); h1 = f2bf(f.y); h2 = f2bf(f.z); h3 = f2bf(f.w);
      }
      uint2 pk;
      pk.x = (unsigned)h0 | ((unsigned)h1 << 16);
      pk.y = (unsigned)h2 | ((unsigned)h3 << 16);
      *(uint2*)&As[r * 32 + k4] = pk;
    }
    #pragma unroll
    for (int it = 0; it < 4; it++) {
      int i = tid + it * 128;            // 0..511
      int kk = i >> 4;                   // 0..31
      int n4 = (i & 15) << 2;            // 0..60
      float4 f = *(const float4*)(W + (size_t)(k0 + kk) * N + n0 + n4);
      BsT[(n4 + 0) * 32 + kk] = f2bf(f.x);
      BsT[(n4 + 1) * 32 + kk] = f2bf(f.y);
      BsT[(n4 + 2) * 32 + kk] = f2bf(f.z);
      BsT[(n4 + 3) * 32 + kk] = f2bf(f.w);
    }
    __syncthreads();
    const ushort_t* arow = &As[(wv * 16 + ln) * 32];
    v16bf a = frag_ld(arow + hi * 8, arow + 16 + hi * 8);
    #pragma unroll
    for (int nt = 0; nt < 4; nt++) {
      const ushort_t* brow = &BsT[(nt * 16 + ln) * 32];
      v16bf b = frag_ld(brow + hi * 8, brow + 16 + hi * 8);
      acc[nt] = wmma_bf16(a, b, acc[nt]);
    }
    __syncthreads();
  }
  #pragma unroll
  for (int nt = 0; nt < 4; nt++) {
    int col = n0 + nt * 16 + ln;
    #pragma unroll
    for (int r = 0; r < 8; r++) {
      int row = m0 + wv * 16 + r + hi * 8;
      if (row < M) Ob[(size_t)row * N + col] = f2bf(acc[nt][r]);
    }
  }
}

// ---------- big GEMM: out[M,N] = A_bf16[M,K] * Bt_bf16[N,K]^T (+bias) ----------
// Block 256 threads (8 waves), tile 128(M) x 128(N), K step 32.
// A and Bt share the [row][k] layout, so staging is a pure async global->LDS copy.
// Requires M%128==0, N%128==0, K%32==0.
template<bool OUT_F32, bool BIAS>
__global__ __launch_bounds__(256)
void gemm_big_wmma(const ushort_t* __restrict__ A, const ushort_t* __restrict__ Bt,
                   const float* __restrict__ bias,
                   float* __restrict__ Of, ushort_t* __restrict__ Ob,
                   int M, int K, int N) {
  __shared__ ushort_t As[128 * 32];   // [m][k]
  __shared__ ushort_t Bs[128 * 32];   // [n][k]
  const int tid = threadIdx.x;
  const int wid = tid >> 5, lane = tid & 31;
  const int ln = lane & 15, hi = lane >> 4;
  const int m0 = blockIdx.y * 128, n0 = blockIdx.x * 128;
  const int wm = wid & 3, wn = wid >> 2;   // 4x2 wave grid: 32(M) x 64(N) per wave

  v8f zero = {};
  v8f acc[2][4] = {{zero, zero, zero, zero}, {zero, zero, zero, zero}};

  // per-lane staging coordinates: 512 chunks of 16B per matrix, 2 issues/thread
  const int r0 = tid >> 1;                 // 0..127 (it advances by +... see loop)
  (void)r0;

  for (int k0 = 0; k0 < K; k0 += 32) {
    #pragma unroll
    for (int it = 0; it < 2; it++) {
      int i = tid + it * 256;              // 0..511
      int r = i >> 2;                      // 0..127
      int c = (i & 3) << 3;                // 0,8,16,24 (elements)
      async_ld_b128((unsigned)(size_t)&As[r * 32 + c], A  + (size_t)(m0 + r) * K + k0 + c);
      async_ld_b128((unsigned)(size_t)&Bs[r * 32 + c], Bt + (size_t)(n0 + r) * K + k0 + c);
    }
    wait_async0();
    __syncthreads();

    v16bf af[2], bf[4];
    #pragma unroll
    for (int mt = 0; mt < 2; mt++) {
      const ushort_t* arow = &As[(wm * 32 + mt * 16 + ln) * 32];
      af[mt] = frag_ld(arow + hi * 8, arow + 16 + hi * 8);
    }
    #pragma unroll
    for (int nt = 0; nt < 4; nt++) {
      const ushort_t* brow = &Bs[(wn * 64 + nt * 16 + ln) * 32];
      bf[nt] = frag_ld(brow + hi * 8, brow + 16 + hi * 8);
    }
    #pragma unroll
    for (int mt = 0; mt < 2; mt++)
      #pragma unroll
      for (int nt = 0; nt < 4; nt++)
        acc[mt][nt] = wmma_bf16(af[mt], bf[nt], acc[mt][nt]);
    __syncthreads();
  }

  #pragma unroll
  for (int mt = 0; mt < 2; mt++) {
    #pragma unroll
    for (int nt = 0; nt < 4; nt++) {
      int col = n0 + wn * 64 + nt * 16 + ln;
      float bv = BIAS ? bias[col] : 0.0f;
      #pragma unroll
      for (int r = 0; r < 8; r++) {
        int row = m0 + wm * 32 + mt * 16 + r + hi * 8;
        float v = acc[mt][nt][r] + bv;
        if (OUT_F32) Of[(size_t)row * N + col] = v;
        else         Ob[(size_t)row * N + col] = f2bf(v);
      }
    }
  }
}

// ---------- antialiased bilinear 512x512 -> 64x64 (JAX-style triangle kernel, radius 8) ----------
__global__ void mask_resize_kernel(const float* __restrict__ src, float* __restrict__ dst) {
  int o = blockIdx.x * blockDim.x + threadIdx.x;
  if (o >= 64 * 64) return;
  int oi = o >> 6, oj = o & 63;
  float acc = 0.f;
  #pragma unroll
  for (int a = 0; a < 16; a++) {
    int ii = oi * 8 - 4 + a; ii = ii < 0 ? 0 : (ii > 511 ? 511 : ii);
    float wa = (1.0f - fabsf((float)a - 7.5f) * 0.125f) * 0.125f;
    float rowacc = 0.f;
    for (int b = 0; b < 16; b++) {
      int jj = oj * 8 - 4 + b; jj = jj < 0 ? 0 : (jj > 511 ? 511 : jj);
      float wb = (1.0f - fabsf((float)b - 7.5f) * 0.125f) * 0.125f;
      rowacc += wb * src[ii * 512 + jj];
    }
    acc += wa * rowacc;
  }
  dst[o] = acc;
}

// ---------- fused dual-branch attention + region-mask combine ----------
// Grid: (S/64, H, B). Block: 128 threads = 4 waves; each wave owns 16 query rows.
#define ATTN_SMEM_USHORTS (96*160 + 160*96 + 16*160 + 160*32 + 4*16*96 + 4*16*32)
#define ATTN_SMEM_BYTES   (ATTN_SMEM_USHORTS * 2)

__global__ __launch_bounds__(128)
void attn_ipadapter_kernel(const ushort_t* __restrict__ Qg, const ushort_t* __restrict__ Kg,
                           const ushort_t* __restrict__ Vg, const ushort_t* __restrict__ ipKg,
                           const ushort_t* __restrict__ ipVg, const float* __restrict__ maskg,
                           ushort_t* __restrict__ Og) {
  extern __shared__ ushort_t smem[];
  ushort_t* Kl    = smem;                 // [96][160]
  ushort_t* VlT   = Kl + 96 * 160;        // [160][96]
  ushort_t* ipKl  = VlT + 160 * 96;       // [16][160]
  ushort_t* ipVlT = ipKl + 16 * 160;      // [160][32]
  ushort_t* Pl    = ipVlT + 160 * 32;     // [4][16][96]
  ushort_t* P2l   = Pl + 4 * 16 * 96;     // [4][16][32]

  const int tid = threadIdx.x;
  const int b = blockIdx.z, h = blockIdx.y, qt = blockIdx.x;

  for (int i = tid; i < 96 * 40; i += 128) {
    int l = i / 40, d4 = (i % 40) * 4;
    uint2 v; v.x = 0u; v.y = 0u;
    if (l < 77) v = *(const uint2*)(Kg + (size_t)(b * 77 + l) * 1280 + h * 160 + d4);
    *(uint2*)&Kl[l * 160 + d4] = v;
  }
  for (int i = tid; i < 96 * 40; i += 128) {
    int l = i / 40, d4 = (i % 40) * 4;
    unsigned x = 0u, y = 0u;
    if (l < 77) { uint2 v = *(const uint2*)(Vg + (size_t)(b * 77 + l) * 1280 + h * 160 + d4); x = v.x; y = v.y; }
    VlT[(d4 + 0) * 96 + l] = (ushort_t)(x & 0xffffu);
    VlT[(d4 + 1) * 96 + l] = (ushort_t)(x >> 16);
    VlT[(d4 + 2) * 96 + l] = (ushort_t)(y & 0xffffu);
    VlT[(d4 + 3) * 96 + l] = (ushort_t)(y >> 16);
  }
  for (int i = tid; i < 16 * 40; i += 128) {
    int l = i / 40, d4 = (i % 40) * 4;
    uint2 v = *(const uint2*)(ipKg + (size_t)(b * 16 + l) * 1280 + h * 160 + d4);
    *(uint2*)&ipKl[l * 160 + d4] = v;
  }
  for (int i = tid; i < 32 * 40; i += 128) {
    int l = i / 40, d4 = (i % 40) * 4;
    unsigned x = 0u, y = 0u;
    if (l < 16) { uint2 v = *(const uint2*)(ipVg + (size_t)(b * 16 + l) * 1280 + h * 160 + d4); x = v.x; y = v.y; }
    ipVlT[(d4 + 0) * 32 + l] = (ushort_t)(x & 0xffffu);
    ipVlT[(d4 + 1) * 32 + l] = (ushort_t)(x >> 16);
    ipVlT[(d4 + 2) * 32 + l] = (ushort_t)(y & 0xffffu);
    ipVlT[(d4 + 3) * 32 + l] = (ushort_t)(y >> 16);
  }
  __syncthreads();

  const int wv = tid >> 5, lane = tid & 31;
  const int ln = lane & 15, hi = lane >> 4;
  const int srow0 = qt * 64 + wv * 16;
  const size_t qrow0 = (size_t)b * 4096 + srow0;

  v16bf qf[5];
  {
    const ushort_t* qbase = Qg + (qrow0 + ln) * 1280 + h * 160;
    #pragma unroll
    for (int ks = 0; ks < 5; ks++)
      qf[ks] = frag_ld(qbase + ks * 32 + hi * 8, qbase + ks * 32 + 16 + hi * 8);
  }

  const float scale = 0.07905694150420949f;  // 1/sqrt(160)
  v8f zero = {};
  v8f sc[5];
  #pragma unroll
  for (int nt = 0; nt < 5; nt++) {
    v8f a = zero;
    #pragma unroll
    for (int ks = 0; ks < 5; ks++) {
      const ushort_t* krow = &Kl[(nt * 16 + ln) * 160 + ks * 32];
      v16bf bf = frag_ld(krow + hi * 8, krow + 16 + hi * 8);
      a = wmma_bf16(qf[ks], bf, a);
    }
    sc[nt] = a;
  }
  #pragma unroll
  for (int nt = 0; nt < 5; nt++) {
    bool invalid = (nt * 16 + ln) >= 77;
    #pragma unroll
    for (int r = 0; r < 8; r++) sc[nt][r] = invalid ? -1.0e30f : sc[nt][r] * scale;
  }
  v8f sc2 = zero;
  #pragma unroll
  for (int ks = 0; ks < 5; ks++) {
    const ushort_t* krow = &ipKl[ln * 160 + ks * 32];
    v16bf bf = frag_ld(krow + hi * 8, krow + 16 + hi * 8);
    sc2 = wmma_bf16(qf[ks], bf, sc2);
  }
  #pragma unroll
  for (int r = 0; r < 8; r++) sc2[r] *= scale;

  float rinv[8], rinv2[8];
  #pragma unroll
  for (int r = 0; r < 8; r++) {
    float m = sc[0][r];
    #pragma unroll
    for (int nt = 1; nt < 5; nt++) m = fmaxf(m, sc[nt][r]);
    #pragma unroll
    for (int s = 8; s >= 1; s >>= 1) m = fmaxf(m, __shfl_xor(m, s, 32));
    float ssum = 0.f;
    #pragma unroll
    for (int nt = 0; nt < 5; nt++) { float e = __expf(sc[nt][r] - m); sc[nt][r] = e; ssum += e; }
    #pragma unroll
    for (int s = 8; s >= 1; s >>= 1) ssum += __shfl_xor(ssum, s, 32);
    rinv[r] = 1.0f / ssum;

    float m2 = sc2[r];
    #pragma unroll
    for (int s = 8; s >= 1; s >>= 1) m2 = fmaxf(m2, __shfl_xor(m2, s, 32));
    float e2 = __expf(sc2[r] - m2); sc2[r] = e2;
    float s2 = e2;
    #pragma unroll
    for (int s = 8; s >= 1; s >>= 1) s2 += __shfl_xor(s2, s, 32);
    rinv2[r] = 1.0f / s2;
  }

  ushort_t* myP  = Pl + wv * 16 * 96;
  ushort_t* myP2 = P2l + wv * 16 * 32;
  #pragma unroll
  for (int r = 0; r < 8; r++) {
    int row = r + hi * 8;
    #pragma unroll
    for (int nt = 0; nt < 5; nt++) myP[row * 96 + nt * 16 + ln] = f2bf(sc[nt][r] * rinv[r]);
    myP[row * 96 + 80 + ln] = 0;
    myP2[row * 32 + ln] = f2bf(sc2[r] * rinv2[r]);
    myP2[row * 32 + 16 + ln] = 0;
  }
  v16bf pf[3], p2f;
  #pragma unroll
  for (int ks = 0; ks < 3; ks++) {
    const ushort_t* prow = &myP[ln * 96 + ks * 32];
    pf[ks] = frag_ld(prow + hi * 8, prow + 16 + hi * 8);
  }
  p2f = frag_ld(&myP2[ln * 32 + hi * 8], &myP2[ln * 32 + 16 + hi * 8]);

  float mk[8];
  #pragma unroll
  for (int r = 0; r < 8; r++) mk[r] = maskg[srow0 + r + hi * 8];

  #pragma unroll
  for (int dt = 0; dt < 10; dt++) {
    v8f o = zero;
    {
      const ushort_t* vrow = &ipVlT[(dt * 16 + ln) * 32];
      v16bf bf = frag_ld(vrow + hi * 8, vrow + 16 + hi * 8);
      o = wmma_bf16(p2f, bf, o);
    }
    #pragma unroll
    for (int r = 0; r < 8; r++) o[r] *= mk[r];
    #pragma unroll
    for (int ks = 0; ks < 3; ks++) {
      const ushort_t* vrow = &VlT[(dt * 16 + ln) * 96 + ks * 32];
      v16bf bf = frag_ld(vrow + hi * 8, vrow + 16 + hi * 8);
      o = wmma_bf16(pf[ks], bf, o);
    }
    #pragma unroll
    for (int r = 0; r < 8; r++)
      Og[(qrow0 + r + hi * 8) * 1280 + h * 160 + dt * 16 + ln] = f2bf(o[r]);
  }
}

// ---------- host launcher ----------
extern "C" void kernel_launch(void* const* d_in, const int* in_sizes, int n_in,
                              void* d_out, int out_size, void* d_ws, size_t ws_size,
                              hipStream_t stream) {
  (void)in_sizes; (void)n_in; (void)out_size; (void)ws_size;
  const float* x    = (const float*)d_in[0];
  const float* ctx  = (const float*)d_in[1];
  const float* iemb = (const float*)d_in[2];
  const float* rmsk = (const float*)d_in[3];
  const float* Wq   = (const float*)d_in[4];
  const float* Wk   = (const float*)d_in[5];
  const float* Wv   = (const float*)d_in[6];
  const float* Wo   = (const float*)d_in[7];
  const float* bo   = (const float*)d_in[8];
  const float* Wik  = (const float*)d_in[9];
  const float* Wiv  = (const float*)d_in[10];
  float* out = (float*)d_out;

  const int B = 4, S = 4096, D = 1280, CD = 768, H = 8;
  const int M = B * S;        // 16384
  const int Mkv = B * 77;     // 308
  const int Mip = B * 16;     // 64

  char* ws = (char*)d_ws;
  size_t off = 0;
  auto suballoc = [&](size_t bytes) -> char* {
    char* p = ws + off;
    off = (off + bytes + 255) & ~(size_t)255;
    return p;
  };
  ushort_t* xb  = (ushort_t*)suballoc((size_t)M * D * 2);     // x in bf16
  ushort_t* WqT = (ushort_t*)suballoc((size_t)D * D * 2);     // Wq^T bf16 [N][K]
  ushort_t* WoT = (ushort_t*)suballoc((size_t)D * D * 2);     // Wo^T bf16 [N][K]
  ushort_t* Qb  = (ushort_t*)suballoc((size_t)M * D * 2);     // Q projection, bf16
  ushort_t* Kb  = (ushort_t*)suballoc((size_t)Mkv * D * 2);   // K projection
  ushort_t* Vb  = (ushort_t*)suballoc((size_t)Mkv * D * 2);   // V projection
  ushort_t* iKb = (ushort_t*)suballoc((size_t)Mip * D * 2);   // ip K
  ushort_t* iVb = (ushort_t*)suballoc((size_t)Mip * D * 2);   // ip V
  ushort_t* Ab  = (ushort_t*)suballoc((size_t)M * D * 2);     // fused attention output
  float*    mk  = (float*)   suballoc(4096 * sizeof(float));  // resized region mask

  // ---- precision/layout prep (bandwidth-trivial) ----
  cvt_f32_bf16_kernel<<<dim3((M * D) / (256 * 8)), dim3(256), 0, stream>>>(x, xb, M * D);
  wt_transpose_cvt_kernel<<<dim3(D / 32, D / 32), dim3(32, 8), 0, stream>>>(Wq, WqT, D, D);
  wt_transpose_cvt_kernel<<<dim3(D / 32, D / 32), dim3(32, 8), 0, stream>>>(Wo, WoT, D, D);

  // ---- small K/V projections (f32 A, bounds-checked) ----
  dim3 blk128(128);
  gemm_small_wmma<<<dim3(D / 64, (Mkv + 63) / 64), blk128, 0, stream>>>(ctx,  Wk,  Kb,  Mkv, CD, D);
  gemm_small_wmma<<<dim3(D / 64, (Mkv + 63) / 64), blk128, 0, stream>>>(ctx,  Wv,  Vb,  Mkv, CD, D);
  gemm_small_wmma<<<dim3(D / 64, (Mip + 63) / 64), blk128, 0, stream>>>(iemb, Wik, iKb, Mip, CD, D);
  gemm_small_wmma<<<dim3(D / 64, (Mip + 63) / 64), blk128, 0, stream>>>(iemb, Wiv, iVb, Mip, CD, D);

  // ---- region mask resize ----
  mask_resize_kernel<<<dim3(16), dim3(256), 0, stream>>>(rmsk, mk);

  // ---- Q projection: big async-staged GEMM (M=16384, N=K=1280) ----
  gemm_big_wmma<false, false><<<dim3(D / 128, M / 128), dim3(256), 0, stream>>>(
      xb, WqT, nullptr, nullptr, Qb, M, D, D);

  // ---- fused dual-branch attention ----
  hipFuncSetAttribute((const void*)attn_ipadapter_kernel,
                      hipFuncAttributeMaxDynamicSharedMemorySize, ATTN_SMEM_BYTES);
  attn_ipadapter_kernel<<<dim3(S / 64, H, B), blk128, ATTN_SMEM_BYTES, stream>>>(
      Qb, Kb, Vb, iKb, iVb, mk, Ab);

  // ---- output projection + bias: big async-staged GEMM, f32 out ----
  gemm_big_wmma<true, true><<<dim3(D / 128, M / 128), dim3(256), 0, stream>>>(
      Ab, WoT, bo, out, nullptr, M, D, D);
}
